// Actor_24721831756419
// MI455X (gfx1250) — matrix operand order repaired and checked by reading, not compiled, not run on hardware
//
#include <hip/hip_runtime.h>

typedef unsigned short u16;
typedef __bf16 v16bf __attribute__((ext_vector_type(16)));
typedef float v8f __attribute__((ext_vector_type(8)));

union Frag {
    v16bf v;
    uint4 q[2];
};

__device__ __forceinline__ float bf2f(u16 h) {
    union { unsigned u; float f; } c;
    c.u = ((unsigned)h) << 16;
    return c.f;
}

__device__ __forceinline__ u16 f2bf(float f) {
    union { float f; unsigned u; } c;
    c.f = f;
    unsigned u = c.u;
    u += 0x7fffu + ((u >> 16) & 1u);   // round-to-nearest-even
    return (u16)(u >> 16);
}

__device__ __forceinline__ float elu(float v) {
    return v > 0.f ? v : (__expf(v) - 1.f);
}

// ---------------------------------------------------------------------------
// Pack all weights f32 -> bf16, transposed to [out][in] so each lane's
// B-fragment is two contiguous 16-byte loads. Wmu out-dim padded 17->32.
//   u16 offsets in ws:
//     [      0 .. 262143]  W1t  [4][256][256]
//     [ 262144 .. 393215]  W2t  [4][128][256]
//     [ 393216 .. 409599]  Wmut [4][ 32][128]
//     [ 409600 .. 417791]  gW1t [32][256]
//     [ 417792 .. 418815]  gW2t [32][32]
// ---------------------------------------------------------------------------
__global__ __launch_bounds__(256) void pack_weights(
    const float* __restrict__ W1, const float* __restrict__ W2,
    const float* __restrict__ Wmu, const float* __restrict__ gW1,
    const float* __restrict__ gW2, u16* __restrict__ ws)
{
    int i = blockIdx.x * 256 + threadIdx.x;
    if (i < 262144) {                         // W1t[e][n][k] = W1[e][k][n]
        int e = i >> 16, n = (i >> 8) & 255, k = i & 255;
        ws[i] = f2bf(W1[(e << 16) + (k << 8) + n]);
    } else if (i < 393216) {                  // W2t[e][n][k] = W2[e][k][n]
        int j = i - 262144;
        int e = j >> 15, n = (j >> 8) & 127, k = j & 255;
        ws[i] = f2bf(W2[(e << 15) + (k << 7) + n]);
    } else if (i < 409600) {                  // Wmut[e][n][k] = Wmu[e][k][n]
        int j = i - 393216;
        int e = j >> 12, n = (j >> 7) & 31, k = j & 127;
        float v = (n < 17) ? Wmu[e * 2176 + k * 17 + n] : 0.f;
        ws[i] = f2bf(v);
    } else if (i < 417792) {                  // gW1t[n][k] = gW1[k][n]
        int j = i - 409600;
        int n = j >> 8, k = j & 255;
        ws[i] = f2bf(gW1[(k << 5) + n]);
    } else if (i < 418816) {                  // gW2t[n][k] = gW2[k][n]
        int j = i - 417792;
        int n = j >> 5, k = j & 31;
        ws[i] = f2bf(gW2[(k << 5) + n]);
    }
}

// ---------------------------------------------------------------------------
// Fused MoE actor, 32 rows per block, 8 waves. All five GEMMs on WMMA.
// ---------------------------------------------------------------------------
#define TM 32

__global__ __launch_bounds__(256) void moe_actor(
    const float* __restrict__ x,
    const float* __restrict__ b1, const float* __restrict__ b2,
    const float* __restrict__ bmu,
    const float* __restrict__ gb1, const float* __restrict__ gb2,
    const float* __restrict__ gWo, const float* __restrict__ gbo,
    const u16* __restrict__ ws, float* __restrict__ out)
{
    __shared__ u16   xs [TM * 256];   // bf16 x tile          16 KB
    __shared__ u16   hs [TM * 256];   // bf16 layer-1 output  16 KB
    __shared__ u16   h2s[TM * 128];   // bf16 layer-2 output   8 KB
    __shared__ u16   g1s[TM * 32];    // bf16 gate h1          2 KB
    __shared__ u16   g2s[TM * 32];    // bf16 gate h2          2 KB
    __shared__ float cs [TM * 4];     // gate coeffs          .5 KB

    const int tid  = threadIdx.x;
    const int wave = tid >> 5;
    const int lane = tid & 31;
    const int half = lane >> 4;      // 16-lane half of the wave
    const int ln   = lane & 15;      // M (A-frag) / N (B,C-frag) index

    // ---- Phase A: stage x tile into LDS as bf16 ---------------------------
    const float* xg = x + (size_t)blockIdx.x * (TM * 256);
    for (int i = tid; i < TM * 256; i += 256) xs[i] = f2bf(xg[i]);
    __syncthreads();

    // ---- Phase B1: gate layer 1 via WMMA  [32,256]x[256,32] ---------------
    if (wave < 4) {
        const u16* gW1t = ws + 409600;         // [32][256]
        int mtile = wave >> 1, otile = wave & 1;
        int arow = (mtile << 4) + ln;
        int col  = (otile << 4) + ln;
        const u16* brow = gW1t + (col << 8);
        v8f acc = {};
        #pragma unroll
        for (int kb = 0; kb < 256; kb += 32) {
            Frag a, b;
            a.q[0] = *(const uint4*)&xs[(arow << 8) + kb + (half << 3)];
            a.q[1] = *(const uint4*)&xs[(arow << 8) + kb + 16 + (half << 3)];
            b.q[0] = *(const uint4*)&brow[kb + (half << 3)];
            b.q[1] = *(const uint4*)&brow[kb + 16 + (half << 3)];
            acc = __builtin_amdgcn_wmma_f32_16x16x32_bf16(
                false, a.v, false, b.v, (short)0, acc, false, false);
        }
        float bias = gb1[col];
        #pragma unroll
        for (int r = 0; r < 8; ++r) {
            int row = (mtile << 4) + (half << 3) + r;
            g1s[(row << 5) + col] = f2bf(elu(acc[r] + bias));
        }
    }
    __syncthreads();

    // ---- Phase B2: gate layer 2 via WMMA  [32,32]x[32,32] -----------------
    if (wave < 4) {
        const u16* gW2t = ws + 417792;         // [32][32]
        int mtile = wave >> 1, otile = wave & 1;
        int arow = (mtile << 4) + ln;
        int col  = (otile << 4) + ln;
        const u16* brow = gW2t + (col << 5);
        Frag a, b;
        a.q[0] = *(const uint4*)&g1s[(arow << 5) + (half << 3)];
        a.q[1] = *(const uint4*)&g1s[(arow << 5) + 16 + (half << 3)];
        b.q[0] = *(const uint4*)&brow[(half << 3)];
        b.q[1] = *(const uint4*)&brow[16 + (half << 3)];
        v8f acc = {};
        acc = __builtin_amdgcn_wmma_f32_16x16x32_bf16(
            false, a.v, false, b.v, (short)0, acc, false, false);
        float bias = gb2[col];
        #pragma unroll
        for (int r = 0; r < 8; ++r) {
            int row = (mtile << 4) + (half << 3) + r;
            g2s[(row << 5) + col] = f2bf(elu(acc[r] + bias));
        }
    }
    __syncthreads();

    // ---- Phase B3: logits + softmax (tiny, VALU) --------------------------
    if (tid < TM * 4) {
        int row = tid >> 2, e = tid & 3;
        float acc = gbo[e];
        for (int k = 0; k < 32; ++k)
            acc = fmaf(bf2f(g2s[(row << 5) + k]), gWo[(k << 2) + e], acc);
        cs[tid] = acc;
    }
    __syncthreads();
    if (tid < TM) {
        float v0 = cs[tid * 4 + 0], v1 = cs[tid * 4 + 1];
        float v2 = cs[tid * 4 + 2], v3 = cs[tid * 4 + 3];
        float m = fmaxf(fmaxf(v0, v1), fmaxf(v2, v3));
        float e0 = __expf(v0 - m), e1 = __expf(v1 - m);
        float e2 = __expf(v2 - m), e3 = __expf(v3 - m);
        float inv = 1.f / (e0 + e1 + e2 + e3);
        cs[tid * 4 + 0] = e0 * inv; cs[tid * 4 + 1] = e1 * inv;
        cs[tid * 4 + 2] = e2 * inv; cs[tid * 4 + 3] = e3 * inv;
    }
    __syncthreads();

    // ---- Phase C: layer 1  h = relu(sum_e c_e (x @ W1_e + b1_e)) ----------
    // Each wave owns an otile; both mtiles + all 4 experts stay in registers
    // so one B fragment feeds 2 WMMAs and one A fragment feeds 4 experts.
    {
        const u16* W1t = ws;                       // [4][256][256]
        #pragma unroll 1
        for (int ot = wave; ot < 16; ot += 8) {
            const u16* brow0 = W1t + (((0 << 8) + (ot << 4) + ln) << 8);
            const u16* brow1 = W1t + (((1 << 8) + (ot << 4) + ln) << 8);
            const u16* brow2 = W1t + (((2 << 8) + (ot << 4) + ln) << 8);
            const u16* brow3 = W1t + (((3 << 8) + (ot << 4) + ln) << 8);
            v8f acc[2][4];
            #pragma unroll
            for (int mt = 0; mt < 2; ++mt)
                #pragma unroll
                for (int e = 0; e < 4; ++e) acc[mt][e] = (v8f){};
            #pragma unroll
            for (int kb = 0; kb < 256; kb += 32) {
                Frag a0, a1, b;
                a0.q[0] = *(const uint4*)&xs[((0 * 16 + ln) << 8) + kb + (half << 3)];
                a0.q[1] = *(const uint4*)&xs[((0 * 16 + ln) << 8) + kb + 16 + (half << 3)];
                a1.q[0] = *(const uint4*)&xs[((1 * 16 + ln) << 8) + kb + (half << 3)];
                a1.q[1] = *(const uint4*)&xs[((1 * 16 + ln) << 8) + kb + 16 + (half << 3)];
                const u16* browe[4] = {brow0, brow1, brow2, brow3};
                #pragma unroll
                for (int e = 0; e < 4; ++e) {
                    b.q[0] = *(const uint4*)&browe[e][kb + (half << 3)];
                    b.q[1] = *(const uint4*)&browe[e][kb + 16 + (half << 3)];
                    acc[0][e] = __builtin_amdgcn_wmma_f32_16x16x32_bf16(
                        false, a0.v, false, b.v, (short)0, acc[0][e], false, false);
                    acc[1][e] = __builtin_amdgcn_wmma_f32_16x16x32_bf16(
                        false, a1.v, false, b.v, (short)0, acc[1][e], false, false);
                }
            }
            #pragma unroll
            for (int mt = 0; mt < 2; ++mt) {
                v8f hc = {};
                #pragma unroll
                for (int e = 0; e < 4; ++e) {
                    float bias = b1[(e << 8) + (ot << 4) + ln];
                    #pragma unroll
                    for (int r = 0; r < 8; ++r) {
                        int row = (mt << 4) + (half << 3) + r;
                        hc[r] += cs[(row << 2) + e] * (acc[mt][e][r] + bias);
                    }
                }
                #pragma unroll
                for (int r = 0; r < 8; ++r) {
                    int row = (mt << 4) + (half << 3) + r;
                    hs[(row << 8) + (ot << 4) + ln] = f2bf(fmaxf(hc[r], 0.f));
                }
            }
        }
    }
    __syncthreads();

    // ---- Phase D: layer 2  h2 = relu(sum_e c_e (h @ W2_e + b2_e)) ---------
    {
        const u16* W2t = ws + 262144;              // [4][128][256]
        int ot = wave;                             // 8 otiles, one per wave
        const u16* brow0 = W2t + (((0 << 7) + (ot << 4) + ln) << 8);
        const u16* brow1 = W2t + (((1 << 7) + (ot << 4) + ln) << 8);
        const u16* brow2 = W2t + (((2 << 7) + (ot << 4) + ln) << 8);
        const u16* brow3 = W2t + (((3 << 7) + (ot << 4) + ln) << 8);
        v8f acc[2][4];
        #pragma unroll
        for (int mt = 0; mt < 2; ++mt)
            #pragma unroll
            for (int e = 0; e < 4; ++e) acc[mt][e] = (v8f){};
        #pragma unroll
        for (int kb = 0; kb < 256; kb += 32) {
            Frag a0, a1, b;
            a0.q[0] = *(const uint4*)&hs[((0 * 16 + ln) << 8) + kb + (half << 3)];
            a0.q[1] = *(const uint4*)&hs[((0 * 16 + ln) << 8) + kb + 16 + (half << 3)];
            a1.q[0] = *(const uint4*)&hs[((1 * 16 + ln) << 8) + kb + (half << 3)];
            a1.q[1] = *(const uint4*)&hs[((1 * 16 + ln) << 8) + kb + 16 + (half << 3)];
            const u16* browe[4] = {brow0, brow1, brow2, brow3};
            #pragma unroll
            for (int e = 0; e < 4; ++e) {
                b.q[0] = *(const uint4*)&browe[e][kb + (half << 3)];
                b.q[1] = *(const uint4*)&browe[e][kb + 16 + (half << 3)];
                acc[0][e] = __builtin_amdgcn_wmma_f32_16x16x32_bf16(
                    false, a0.v, false, b.v, (short)0, acc[0][e], false, false);
                acc[1][e] = __builtin_amdgcn_wmma_f32_16x16x32_bf16(
                    false, a1.v, false, b.v, (short)0, acc[1][e], false, false);
            }
        }
        #pragma unroll
        for (int mt = 0; mt < 2; ++mt) {
            v8f hc = {};
            #pragma unroll
            for (int e = 0; e < 4; ++e) {
                float bias = b2[(e << 7) + (ot << 4) + ln];
                #pragma unroll
                for (int r = 0; r < 8; ++r) {
                    int row = (mt << 4) + (half << 3) + r;
                    hc[r] += cs[(row << 2) + e] * (acc[mt][e][r] + bias);
                }
            }
            #pragma unroll
            for (int r = 0; r < 8; ++r) {
                int row = (mt << 4) + (half << 3) + r;
                h2s[(row << 7) + (ot << 4) + ln] = f2bf(fmaxf(hc[r], 0.f));
            }
        }
    }
    __syncthreads();

    // ---- Phase E: head  mu = sum_e c_e (h2 @ Wmu_e + bmu_e) ---------------
    if (wave < 4) {                                // 2 mtiles x 2 otiles
        const u16* Wmut = ws + 393216;             // [4][32][128]
        int mtile = wave >> 1, otile = wave & 1;
        int arow = (mtile << 4) + ln;
        int col  = (otile << 4) + ln;
        v8f hc = {};
        #pragma unroll
        for (int e = 0; e < 4; ++e) {
            v8f acc = {};
            const u16* brow = Wmut + (((e << 5) + (otile << 4) + ln) << 7);
            #pragma unroll
            for (int kb = 0; kb < 128; kb += 32) {
                Frag a, b;
                a.q[0] = *(const uint4*)&h2s[(arow << 7) + kb + (half << 3)];
                a.q[1] = *(const uint4*)&h2s[(arow << 7) + kb + 16 + (half << 3)];
                b.q[0] = *(const uint4*)&brow[kb + (half << 3)];
                b.q[1] = *(const uint4*)&brow[kb + 16 + (half << 3)];
                acc = __builtin_amdgcn_wmma_f32_16x16x32_bf16(
                    false, a.v, false, b.v, (short)0, acc, false, false);
            }
            float bias = (col < 17) ? bmu[e * 17 + col] : 0.f;
            #pragma unroll
            for (int r = 0; r < 8; ++r) {
                int row = (mtile << 4) + (half << 3) + r;
                hc[r] += cs[(row << 2) + e] * (acc[r] + bias);
            }
        }
        if (col < 17) {
            #pragma unroll
            for (int r = 0; r < 8; ++r) {
                int row = (mtile << 4) + (half << 3) + r;
                out[((size_t)blockIdx.x * TM + row) * 17 + col] = hc[r];
            }
        }
    }
}

extern "C" void kernel_launch(void* const* d_in, const int* in_sizes, int n_in,
                              void* d_out, int out_size, void* d_ws, size_t ws_size,
                              hipStream_t stream) {
    const float* x   = (const float*)d_in[0];
    const float* W1  = (const float*)d_in[1];
    const float* b1  = (const float*)d_in[2];
    const float* W2  = (const float*)d_in[3];
    const float* b2  = (const float*)d_in[4];
    const float* Wmu = (const float*)d_in[5];
    const float* bmu = (const float*)d_in[6];
    const float* gW1 = (const float*)d_in[7];
    const float* gb1 = (const float*)d_in[8];
    const float* gW2 = (const float*)d_in[9];
    const float* gb2 = (const float*)d_in[10];
    const float* gWo = (const float*)d_in[11];
    const float* gbo = (const float*)d_in[12];
    float* out = (float*)d_out;
    u16*   ws  = (u16*)d_ws;

    // 418816 packed bf16 elements total -> 1636 blocks of 256
    pack_weights<<<1636, 256, 0, stream>>>(W1, W2, Wmu, gW1, gW2, ws);

    int B = in_sizes[0] / 256;          // 131072
    int grid = B / TM;                  // 4096 blocks of 32 rows
    moe_actor<<<grid, 256, 0, stream>>>(x, b1, b2, bmu,
                                        gb1, gb2, gWo, gbo,
                                        ws, out);
}
